// SegGPT_30253749633723
// MI455X (gfx1250) — matrix operand order repaired and checked by reading, not compiled, not run on hardware
//
#include <hip/hip_runtime.h>
#include <math.h>

// ---------------------------------------------------------------------------
// Types / helpers
// ---------------------------------------------------------------------------
typedef __attribute__((ext_vector_type(16))) __bf16 v16bf;
typedef __attribute__((ext_vector_type(8)))  float  v8f;

struct alignas(16) U16x8 { unsigned int x[4]; };       // 8 ushorts / 16 B
struct U16x16 { U16x8 lo, hi; };                       // 16 ushorts / 32 B

__device__ inline unsigned short f2bf(float f) {
  unsigned int u = __builtin_bit_cast(unsigned int, f);
  u += 0x7FFFu + ((u >> 16) & 1u);            // round-to-nearest-even
  return (unsigned short)(u >> 16);
}
__device__ inline float bf2f(unsigned short s) {
  return __builtin_bit_cast(float, (unsigned int)s << 16);
}
__device__ inline float gelu_f(float x) {
  return 0.5f * x * (1.0f + erff(x * 0.70710678118654752f));
}
// build v16bf from two 16B LDS chunks (each a ds_load_b128)
__device__ inline v16bf frag16(const unsigned short* p0, const unsigned short* p1) {
  U16x16 t;
  t.lo = *(const U16x8*)p0;
  t.hi = *(const U16x8*)p1;
  return __builtin_bit_cast(v16bf, t);
}

// --- CDNA5 async global->LDS (per-lane 16B copy, tracked by ASYNCcnt) ------
__device__ inline void async_cp16(const void* gsrc, void* ldst) {
  unsigned int l = (unsigned int)(size_t)ldst;             // low 32b = LDS addr
  unsigned long long g = (unsigned long long)(size_t)gsrc;
  asm volatile("global_load_async_to_lds_b128 %0, %1, off"
               :: "v"(l), "v"(g) : "memory");
}
__device__ inline void wait_async() {
  asm volatile("s_wait_asynccnt 0x0" ::: "memory");
}

// Problem constants
#define RB    4608      // B*H*W tokens
#define CC    768
#define C3    2304
#define NTOK  2304
#define NHEAD 24        // B*HEADS
#define HDIM  64
#define MLPD  3072
#define BOTD  384
#define KC2   3456      // 9*384
#define LPAD  40        // LDS row pitch in ushorts (80B = 20 banks, conflict-free)

// ---------------------------------------------------------------------------
// Epilogue functors
// ---------------------------------------------------------------------------
struct EpiQKV {          // scatter qkv output -> q/k [bh][tok][d], v transposed [bh][d][tok]
  const float* bias;
  unsigned short* q; unsigned short* k; unsigned short* vt;
  __device__ void operator()(int m, int n, float acc) const {
    float val = acc + bias[n];
    int three = n / 768, rem = n % 768;
    int head = rem / 64, d = rem % 64;
    int b = m / NTOK, tok = m % NTOK;
    int bh = b * 12 + head;
    unsigned short h = f2bf(val);
    if (three == 0)      q[(((size_t)bh) * NTOK + tok) * 64 + d] = h;
    else if (three == 1) k[(((size_t)bh) * NTOK + tok) * 64 + d] = h;
    else                 vt[(((size_t)bh) * 64 + d) * NTOK + tok] = h;
  }
};

struct EpiScores {       // attn = acc*scale + rel_h + rel_w  (per-bh pointers)
  const float* relh; const float* relw; float* attn; float scale;
  __device__ void operator()(int m, int n, float acc) const {
    attn[(size_t)m * NTOK + n] =
        acc * scale + relh[m * 48 + (n / 48)] + relw[m * 48 + (n % 48)];
  }
};

struct EpiAV {           // probs@V -> attn_out[b*2304+m][head*64+n] bf16
  unsigned short* out; int b; int head;
  __device__ void operator()(int m, int n, float acc) const {
    out[((size_t)(b * NTOK + m)) * CC + head * 64 + n] = f2bf(acc);
  }
};

struct EpiBiasRes {      // proj: acc + bias + residual -> f32
  const float* bias; const float* res; float* out; int ldc;
  __device__ void operator()(int m, int n, float acc) const {
    out[(size_t)m * ldc + n] = acc + bias[n] + res[(size_t)m * ldc + n];
  }
};

struct EpiGeluBf {       // fc1: gelu(acc + bias) -> bf16
  const float* bias; unsigned short* out; int ldc;
  __device__ void operator()(int m, int n, float acc) const {
    out[(size_t)m * ldc + n] = f2bf(gelu_f(acc + bias[n]));
  }
};

struct EpiFc2 {          // fc2: acc + bias + x1 -> f32 x2 and bf16 x2
  const float* bias; const float* res; float* outf; unsigned short* outb;
  __device__ void operator()(int m, int n, float acc) const {
    float v = acc + bias[n] + res[(size_t)m * CC + n];
    outf[(size_t)m * CC + n] = v;
    outb[(size_t)m * CC + n] = f2bf(v);
  }
};

struct EpiF32 {          // raw conv output -> f32
  float* out; int ldc;
  __device__ void operator()(int m, int n, float acc) const {
    out[(size_t)m * ldc + n] = acc;
  }
};

// ---------------------------------------------------------------------------
// bf16 WMMA GEMM: C[M,N] = A[M,K] * B[N,K]^T   (B stored N-major: Bsrc[n*ldb+k])
//  - 256 threads = 8 waves; wave grid WM x WN (WM*WN==8); wave tile 32x32
//  - block tile (WM*32) x (WN*32); K step 32; all dims divide exactly
//  - double-buffered LDS; tiles DMA'd with global_load_async_to_lds_b128,
//    next tile's async issue overlapped with current tile's WMMAs
//  - CONV2: A rows gathered as 3x3 implicit im2col; OOB rows read a zero page
// ---------------------------------------------------------------------------
template <int WM, int WN, bool CONV2, class Epi>
__global__ __launch_bounds__(256) void gemm_bf16(
    const unsigned short* __restrict__ A, int lda,
    const unsigned short* __restrict__ Bm, int ldb, int K,
    const unsigned short* __restrict__ zpage, Epi epi) {
  constexpr int BM = WM * 32;
  constexpr int BN = WN * 32;
  constexpr int ASEG = (BM * 32) / (256 * 8);
  constexpr int BSEG = (BN * 32) / (256 * 8);
  __shared__ alignas(16) unsigned short sA[2][BM][LPAD];   // [buf][row][k]
  __shared__ alignas(16) unsigned short sBt[2][BN][LPAD];  // [buf][col][k]

  const int t = threadIdx.x;
  const int wave = t >> 5, lane = t & 31;
  const int wm = wave / WN, wn = wave % WN;
  const int m0 = blockIdx.y * BM, n0 = blockIdx.x * BN;

  auto issue = [&](int kk, int p) {
#pragma unroll
    for (int s = 0; s < ASEG; ++s) {
      int e = (s * 256 + t) * 8;
      int r = e >> 5, c = e & 31;
      const unsigned short* g;
      if constexpr (!CONV2) {
        g = A + (size_t)(m0 + r) * lda + kk + c;
      } else {
        int m = m0 + r;
        int kg = kk + c;                 // tap-aligned 8-seg (ci%8==0)
        int tap = kg / BOTD, ci = kg % BOTD;
        int dy = tap / 3 - 1, dx = tap % 3 - 1;
        int b = m / NTOK, rem = m % NTOK;
        int y = rem / 48 + dy, x = rem % 48 + dx;
        g = (y >= 0 && y < 48 && x >= 0 && x < 48)
                ? A + ((size_t)(b * NTOK + y * 48 + x)) * BOTD + ci
                : zpage;                  // DMA zeros for padding
      }
      async_cp16(g, &sA[p][r][c]);
    }
#pragma unroll
    for (int s = 0; s < BSEG; ++s) {
      int e = (s * 256 + t) * 8;
      int n = e >> 5, c = e & 31;
      async_cp16(Bm + (size_t)(n0 + n) * ldb + kk + c, &sBt[p][n][c]);
    }
  };

  v8f acc[2][2];
#pragma unroll
  for (int i = 0; i < 2; ++i)
#pragma unroll
    for (int j = 0; j < 2; ++j)
#pragma unroll
      for (int v = 0; v < 8; ++v) acc[i][j][v] = 0.0f;

  issue(0, 0);
  int p = 0;
  for (int kk = 0; kk < K; kk += 32, p ^= 1) {
    wait_async();        // this wave's async copies into buffer p complete
    __syncthreads();     // all waves' copies complete
    if (kk + 32 < K) issue(kk + 32, p ^ 1);  // overlap DMA with WMMAs

    // ---- fragments (pure b128 LDS reads) + WMMA --------------------------
    const int rowA = lane & 15, hi = lane >> 4;
    v16bf af[2], bfr[2];
#pragma unroll
    for (int i = 0; i < 2; ++i) {
      const unsigned short* rp = sA[p][wm * 32 + i * 16 + rowA];
      af[i] = frag16(rp + hi * 8, rp + 16 + hi * 8);   // k: hi*8..+7, 16+hi*8..+7
    }
#pragma unroll
    for (int j = 0; j < 2; ++j) {
      const unsigned short* cp = sBt[p][wn * 32 + j * 16 + rowA];
      bfr[j] = frag16(cp + hi * 8, cp + 16 + hi * 8);  // k: hi*8..+7, 16+hi*8..+7
    }
#pragma unroll
    for (int i = 0; i < 2; ++i)
#pragma unroll
      for (int j = 0; j < 2; ++j)
        acc[i][j] = __builtin_amdgcn_wmma_f32_16x16x32_bf16(
            false, af[i], false, bfr[j], (short)0, acc[i][j], false, false);
  }

  // ---- epilogue: lane<16 -> M=v, lane>=16 -> M=v+8; N=lane&15 ------------
  const int nl = lane & 15, mh = (lane >> 4) * 8;
#pragma unroll
  for (int i = 0; i < 2; ++i)
#pragma unroll
    for (int j = 0; j < 2; ++j)
#pragma unroll
      for (int v = 0; v < 8; ++v)
        epi(m0 + wm * 32 + i * 16 + mh + v,
            n0 + wn * 32 + j * 16 + nl, acc[i][j][v]);
}

// ---------------------------------------------------------------------------
// Row LayerNorm (also channel-LN in NHWC) + optional GELU / residual
// ---------------------------------------------------------------------------
__global__ __launch_bounds__(256) void ln_kernel(
    const float* __restrict__ in, const float* __restrict__ resid,
    const float* __restrict__ w, const float* __restrict__ bb,
    unsigned short* outb, float* outf, int C, float eps, int do_gelu) {
  __shared__ float s1[256], s2[256];
  const int row = blockIdx.x;
  const float* rp = in + (size_t)row * C;
  float sum = 0.f, sq = 0.f;
  for (int c = threadIdx.x; c < C; c += 256) {
    float v = rp[c]; sum += v; sq += v * v;
  }
  s1[threadIdx.x] = sum; s2[threadIdx.x] = sq;
  __syncthreads();
  for (int off = 128; off; off >>= 1) {
    if (threadIdx.x < off) {
      s1[threadIdx.x] += s1[threadIdx.x + off];
      s2[threadIdx.x] += s2[threadIdx.x + off];
    }
    __syncthreads();
  }
  float mean = s1[0] / (float)C;
  float var  = s2[0] / (float)C - mean * mean;
  float rstd = rsqrtf(var + eps);
  for (int c = threadIdx.x; c < C; c += 256) {
    float v = (rp[c] - mean) * rstd * w[c] + bb[c];
    if (do_gelu) v = gelu_f(v);
    if (outb) outb[(size_t)row * C + c] = f2bf(v);
    if (outf) outf[(size_t)row * C + c] =
        v + (resid ? resid[(size_t)row * C + c] : 0.f);
  }
}

// ---------------------------------------------------------------------------
// Softmax over rows of 2304, f32 in -> bf16 probs out
// ---------------------------------------------------------------------------
__global__ __launch_bounds__(256) void softmax_kernel(
    const float* __restrict__ in, unsigned short* __restrict__ out) {
  __shared__ float sh[256];
  const int row = blockIdx.x, t = threadIdx.x;
  const float* rp = in + (size_t)row * NTOK;
  float vv[9];
  float mx = -1e30f;
#pragma unroll
  for (int i = 0; i < 9; ++i) { vv[i] = rp[t + i * 256]; mx = fmaxf(mx, vv[i]); }
  sh[t] = mx; __syncthreads();
  for (int off = 128; off; off >>= 1) {
    if (t < off) sh[t] = fmaxf(sh[t], sh[t + off]);
    __syncthreads();
  }
  mx = sh[0]; __syncthreads();
  float sum = 0.f;
#pragma unroll
  for (int i = 0; i < 9; ++i) { vv[i] = expf(vv[i] - mx); sum += vv[i]; }
  sh[t] = sum; __syncthreads();
  for (int off = 128; off; off >>= 1) {
    if (t < off) sh[t] += sh[t + off];
    __syncthreads();
  }
  float inv = 1.f / sh[0];
#pragma unroll
  for (int i = 0; i < 9; ++i)
    out[(size_t)row * NTOK + t + i * 256] = f2bf(vv[i] * inv);
}

// ---------------------------------------------------------------------------
// Decomposed rel-pos: rel[bh,tok,k] = sum_d q[bh,tok,d] * table[pos-k+47, d]
// mode 0: pos = tok/48 (h);  mode 1: pos = tok%48 (w)
// ---------------------------------------------------------------------------
__global__ __launch_bounds__(256) void rel_kernel(
    const unsigned short* __restrict__ q, const float* __restrict__ table,
    float* __restrict__ out, int mode, int total) {
  int i = blockIdx.x * 256 + threadIdx.x;
  if (i >= total) return;
  int k = i % 48;
  int rem = i / 48;
  int tok = rem % NTOK;
  int bh = rem / NTOK;
  int pos = mode ? (tok % 48) : (tok / 48);
  const float* tr = table + (size_t)(pos - k + 47) * HDIM;
  const U16x8* qp = (const U16x8*)(q + ((size_t)bh * NTOK + tok) * HDIM);
  float s = 0.f;
#pragma unroll
  for (int g = 0; g < 8; ++g) {
    U16x8 u = qp[g];
    const unsigned short* us = (const unsigned short*)&u;
#pragma unroll
    for (int j = 0; j < 8; ++j) s += bf2f(us[j]) * tr[g * 8 + j];
  }
  out[i] = s;
}

// ---------------------------------------------------------------------------
// Weight conversion / repack (all B operands become N-major bf16)
// ---------------------------------------------------------------------------
__global__ __launch_bounds__(256) void cvt_bf16_kernel(
    const float* __restrict__ src, unsigned short* __restrict__ dst, int n) {
  int i = blockIdx.x * 256 + threadIdx.x;
  if (i < n) dst[i] = f2bf(src[i]);
}

// src [Kdim][Ndim] row-major -> dst [Ndim][Kdim] (N-major) bf16
__global__ __launch_bounds__(256) void cvt_tr_kernel(
    const float* __restrict__ src, unsigned short* __restrict__ dst,
    int Kdim, int Ndim) {
  int i = blockIdx.x * 256 + threadIdx.x;
  if (i >= Kdim * Ndim) return;
  int n = i / Kdim, k = i % Kdim;
  dst[i] = f2bf(src[(size_t)k * Ndim + n]);
}

// conv2_w (O=384, I=384, 3, 3) -> dst[o][tap*384+ci]  (N-major, K=3456)
__global__ __launch_bounds__(256) void repack_c2_kernel(
    const float* __restrict__ src, unsigned short* __restrict__ dst, int n) {
  int i = blockIdx.x * 256 + threadIdx.x;
  if (i >= n) return;
  int o = i / KC2;
  int r = i % KC2;
  int tap = r / BOTD, ci = r % BOTD;
  dst[i] = f2bf(src[((size_t)o * BOTD + ci) * 9 + tap]);
}

__global__ void zero_kernel(unsigned short* p, int n) {
  int i = blockIdx.x * blockDim.x + threadIdx.x;
  if (i < n) p[i] = 0;
}

// ---------------------------------------------------------------------------
// Host orchestration
// ---------------------------------------------------------------------------
extern "C" void kernel_launch(void* const* d_in, const int* in_sizes, int n_in,
                              void* d_out, int out_size, void* d_ws,
                              size_t ws_size, hipStream_t stream) {
  (void)in_sizes; (void)n_in; (void)out_size; (void)ws_size;
  const float* x       = (const float*)d_in[0];
  const float* norm1_w = (const float*)d_in[1];
  const float* norm1_b = (const float*)d_in[2];
  const float* qkv_w   = (const float*)d_in[3];
  const float* qkv_b   = (const float*)d_in[4];
  const float* proj_w  = (const float*)d_in[5];
  const float* proj_b  = (const float*)d_in[6];
  const float* rph     = (const float*)d_in[7];
  const float* rpw     = (const float*)d_in[8];
  const float* norm2_w = (const float*)d_in[9];
  const float* norm2_b = (const float*)d_in[10];
  const float* fc1_w   = (const float*)d_in[11];
  const float* fc1_b   = (const float*)d_in[12];
  const float* fc2_w   = (const float*)d_in[13];
  const float* fc2_b   = (const float*)d_in[14];
  const float* conv1_w = (const float*)d_in[15];
  const float* rln1_w  = (const float*)d_in[16];
  const float* rln1_b  = (const float*)d_in[17];
  const float* conv2_w = (const float*)d_in[18];
  const float* rln2_w  = (const float*)d_in[19];
  const float* rln2_b  = (const float*)d_in[20];
  const float* conv3_w = (const float*)d_in[21];
  const float* rln3_w  = (const float*)d_in[22];
  const float* rln3_b  = (const float*)d_in[23];
  float* outp = (float*)d_out;

  // Bump allocator over workspace
  char* base = (char*)d_ws;
  size_t off = 0;
  auto alloc = [&](size_t bytes) -> void* {
    void* p = base + off;
    off = (off + bytes + 255) & ~(size_t)255;
    return p;
  };
  const int CH = 4;  // heads processed per attention chunk

  unsigned short* zp   = (unsigned short*)alloc(256);   // DMA zero page
  unsigned short* xn   = (unsigned short*)alloc((size_t)RB * CC * 2);
  unsigned short* qb   = (unsigned short*)alloc((size_t)NHEAD * NTOK * HDIM * 2);
  unsigned short* kb   = (unsigned short*)alloc((size_t)NHEAD * NTOK * HDIM * 2);
  unsigned short* vbt  = (unsigned short*)alloc((size_t)NHEAD * NTOK * HDIM * 2);
  float* relh          = (float*)alloc((size_t)NHEAD * NTOK * 48 * 4);
  float* relw          = (float*)alloc((size_t)NHEAD * NTOK * 48 * 4);
  float* attn          = (float*)alloc((size_t)CH * NTOK * NTOK * 4);
  unsigned short* prob = (unsigned short*)alloc((size_t)CH * NTOK * NTOK * 2);
  unsigned short* aout = (unsigned short*)alloc((size_t)RB * CC * 2);
  float* x1            = (float*)alloc((size_t)RB * CC * 4);
  unsigned short* hdn  = (unsigned short*)alloc((size_t)RB * MLPD * 2);
  float* x2            = (float*)alloc((size_t)RB * CC * 4);
  unsigned short* x2b  = (unsigned short*)alloc((size_t)RB * CC * 2);
  float* o1raw         = (float*)alloc((size_t)RB * BOTD * 4);
  unsigned short* o1b  = (unsigned short*)alloc((size_t)RB * BOTD * 2);
  float* o2raw         = (float*)alloc((size_t)RB * BOTD * 4);
  unsigned short* o2b  = (unsigned short*)alloc((size_t)RB * BOTD * 2);
  float* c3raw         = (float*)alloc((size_t)RB * CC * 4);
  unsigned short* qkvw_b = (unsigned short*)alloc((size_t)CC * C3 * 2);
  unsigned short* projw_b= (unsigned short*)alloc((size_t)CC * CC * 2);
  unsigned short* fc1w_b = (unsigned short*)alloc((size_t)CC * MLPD * 2);
  unsigned short* fc2w_b = (unsigned short*)alloc((size_t)MLPD * CC * 2);
  unsigned short* c1w_b  = (unsigned short*)alloc((size_t)BOTD * CC * 2);
  unsigned short* c2w_b  = (unsigned short*)alloc((size_t)KC2 * BOTD * 2);
  unsigned short* c3w_b  = (unsigned short*)alloc((size_t)CC * BOTD * 2);

  auto blocks = [](int n) { return (n + 255) / 256; };

  // --- zero page + weight conversion (transpose to N-major bf16) ---------
  zero_kernel<<<1, 256, 0, stream>>>(zp, 128);
  cvt_tr_kernel<<<blocks(CC * C3), 256, 0, stream>>>(qkv_w, qkvw_b, CC, C3);
  cvt_tr_kernel<<<blocks(CC * CC), 256, 0, stream>>>(proj_w, projw_b, CC, CC);
  cvt_tr_kernel<<<blocks(CC * MLPD), 256, 0, stream>>>(fc1_w, fc1w_b, CC, MLPD);
  cvt_tr_kernel<<<blocks(MLPD * CC), 256, 0, stream>>>(fc2_w, fc2w_b, MLPD, CC);
  // conv 1x1 OIHW weights are already N(=O)-major in K(=I)
  cvt_bf16_kernel<<<blocks(BOTD * CC), 256, 0, stream>>>(conv1_w, c1w_b, BOTD * CC);
  cvt_bf16_kernel<<<blocks(CC * BOTD), 256, 0, stream>>>(conv3_w, c3w_b, CC * BOTD);
  repack_c2_kernel<<<blocks(BOTD * KC2), 256, 0, stream>>>(conv2_w, c2w_b, BOTD * KC2);

  // --- LN1 -> xn (bf16) --------------------------------------------------
  ln_kernel<<<RB, 256, 0, stream>>>(x, nullptr, norm1_w, norm1_b, xn, nullptr,
                                    CC, 1e-5f, 0);

  // --- QKV GEMM: [4608x768] @ [768x2304] ---------------------------------
  {
    EpiQKV e{qkv_b, qb, kb, vbt};
    gemm_bf16<2, 4, false, EpiQKV>
        <<<dim3(C3 / 128, RB / 64), 256, 0, stream>>>(xn, CC, qkvw_b, CC, CC, zp, e);
  }

  // --- decomposed rel-pos biases -----------------------------------------
  {
    int total = NHEAD * NTOK * 48;
    rel_kernel<<<blocks(total), 256, 0, stream>>>(qb, rph, relh, 0, total);
    rel_kernel<<<blocks(total), 256, 0, stream>>>(qb, rpw, relw, 1, total);
  }

  // --- attention, CH heads at a time (stream-serialized buffer reuse) ----
  const float scale = 0.125f;  // 64^-0.5
  for (int c0 = 0; c0 < NHEAD; c0 += CH) {
    for (int bh = c0; bh < c0 + CH; ++bh) {
      EpiScores e{relh + (size_t)bh * NTOK * 48, relw + (size_t)bh * NTOK * 48,
                  attn + (size_t)(bh - c0) * NTOK * NTOK, scale};
      // B = K matrix, N-major: kb[keytok][d]
      gemm_bf16<2, 4, false, EpiScores>
          <<<dim3(NTOK / 128, NTOK / 64), 256, 0, stream>>>(
              qb + (size_t)bh * NTOK * HDIM, HDIM,
              kb + (size_t)bh * NTOK * HDIM, HDIM, HDIM, zp, e);
    }
    softmax_kernel<<<CH * NTOK, 256, 0, stream>>>(attn, prob);
    for (int bh = c0; bh < c0 + CH; ++bh) {
      EpiAV e{aout, bh / 12, bh % 12};
      // B = V^T, N-major: vbt[d][keytok]
      gemm_bf16<4, 2, false, EpiAV>
          <<<dim3(HDIM / 64, NTOK / 128), 256, 0, stream>>>(
              prob + (size_t)(bh - c0) * NTOK * NTOK, NTOK,
              vbt + (size_t)bh * NTOK * HDIM, NTOK, NTOK, zp, e);
    }
  }

  // --- proj + shortcut residual ------------------------------------------
  {
    EpiBiasRes e{proj_b, x, x1, CC};
    gemm_bf16<2, 4, false, EpiBiasRes>
        <<<dim3(CC / 128, RB / 64), 256, 0, stream>>>(aout, CC, projw_b, CC, CC, zp, e);
  }

  // --- LN2 -> xn (reused) -------------------------------------------------
  ln_kernel<<<RB, 256, 0, stream>>>(x1, nullptr, norm2_w, norm2_b, xn, nullptr,
                                    CC, 1e-5f, 0);

  // --- MLP ----------------------------------------------------------------
  {
    EpiGeluBf e{fc1_b, hdn, MLPD};
    gemm_bf16<2, 4, false, EpiGeluBf>
        <<<dim3(MLPD / 128, RB / 64), 256, 0, stream>>>(xn, CC, fc1w_b, CC, CC, zp, e);
  }
  {
    EpiFc2 e{fc2_b, x1, x2, x2b};
    gemm_bf16<2, 4, false, EpiFc2>
        <<<dim3(CC / 128, RB / 64), 256, 0, stream>>>(hdn, MLPD, fc2w_b, MLPD, MLPD, zp, e);
  }

  // --- bottleneck: conv1 (1x1) -> LN+GELU ---------------------------------
  {
    EpiF32 e{o1raw, BOTD};
    gemm_bf16<2, 4, false, EpiF32>
        <<<dim3(BOTD / 128, RB / 64), 256, 0, stream>>>(x2b, CC, c1w_b, CC, CC, zp, e);
  }
  ln_kernel<<<RB, 256, 0, stream>>>(o1raw, nullptr, rln1_w, rln1_b, o1b, nullptr,
                                    BOTD, 1e-6f, 1);

  // --- conv2 (3x3 implicit im2col GEMM, K=3456) -> LN+GELU ----------------
  {
    EpiF32 e{o2raw, BOTD};
    gemm_bf16<2, 4, true, EpiF32>
        <<<dim3(BOTD / 128, RB / 64), 256, 0, stream>>>(o1b, BOTD, c2w_b, KC2,
                                                        KC2, zp, e);
  }
  ln_kernel<<<RB, 256, 0, stream>>>(o2raw, nullptr, rln2_w, rln2_b, o2b, nullptr,
                                    BOTD, 1e-6f, 1);

  // --- conv3 (1x1) -> LN(rln3) + residual(x2) -> d_out --------------------
  {
    EpiF32 e{c3raw, CC};
    gemm_bf16<2, 4, false, EpiF32>
        <<<dim3(CC / 128, RB / 64), 256, 0, stream>>>(o2b, BOTD, c3w_b, BOTD,
                                                      BOTD, zp, e);
  }
  ln_kernel<<<RB, 256, 0, stream>>>(c3raw, x2, rln3_w, rln3_b, nullptr, outp,
                                    CC, 1e-6f, 0);
}
